// PointTransformerLayer_79946521247960
// MI455X (gfx1250) — compile-verified, matrix operand order checked by reading, not run on hardware
//
#include <hip/hip_runtime.h>
#include <hip/hip_bf16.h>

typedef __attribute__((ext_vector_type(16))) _Float16 v16h;
typedef __attribute__((ext_vector_type(8)))  float    v8f;

#define BATCH 4
#define NPTS  4096
#define KNN   16
#define DIM   64

__device__ __forceinline__ void wait_ds() {
    asm volatile("s_wait_dscnt 0" ::: "memory");
}

// ---- WMMA fragment index helpers (CDNA5 wave32 layouts, CDNA5 ISA) ----
// A (16x32 f16): lane = r + 16*((k%16)>>3), elem = ((k%32)>>4)*8 + (k&7)
__device__ __forceinline__ int fragLane(int r, int k) { return r + (((k & 15) >> 3) << 4); }
__device__ __forceinline__ int fragElem(int k) { int kk = k & 31; return ((kk >> 4) << 3) + (kk & 7); }

__device__ __forceinline__ v8f wmma_f16(v16h a, v16h b, v8f c) {
    return __builtin_amdgcn_wmma_f32_16x16x32_f16(false, a, false, b, (short)0, c, false, false);
}

// B-fragment buffer: per matrix 2(kh) x 4(nt) x 32(lane) x 16(e) halfs = 8KB
__device__ __forceinline__ v16h loadBfrag(const _Float16* wf, int mat, int kh, int nt, int lane) {
    const v16h* p = (const v16h*)(wf + mat * 4096);
    return p[(kh * 4 + nt) * 32 + lane];
}

// =================== weight repack: f32 row-major -> f16 B-fragment order ==========
__global__ __launch_bounds__(256) void prep_kernel(
    const float* m0, const float* m1, const float* m2, const float* m3,
    const float* m4, const float* m5, const float* m6, const float* m7,
    _Float16* wf) {
    int mat = blockIdx.x;
    const float* W = (mat == 0) ? m0 : (mat == 1) ? m1 : (mat == 2) ? m2 : (mat == 3) ? m3
                   : (mat == 4) ? m4 : (mat == 5) ? m5 : (mat == 6) ? m6 : m7;
    int t = threadIdx.x;
    int kh = t >> 7, nt = (t >> 5) & 3, lane = t & 31;
    _Float16* dst = wf + mat * 4096 + t * 16;
#pragma unroll
    for (int e = 0; e < 16; ++e) {
        int k = 32 * kh + ((e >> 3) << 4) + ((lane >> 4) << 3) + (e & 7);
        int n = 16 * nt + (lane & 15);
        dst[e] = (_Float16)W[k * 64 + n];
    }
}

// =================== brute-force KNN, LDS tiled, register top-16 ===================
__global__ __launch_bounds__(256) void knn_kernel(const float* __restrict__ coords,
                                                  int* __restrict__ knn) {
    __shared__ float sc[256 * 3];
    int b = blockIdx.x >> 4;          // 16 blocks per batch
    int qi = ((blockIdx.x & 15) << 8) + threadIdx.x;
    const float* cb = coords + (size_t)b * NPTS * 3;
    float qx = cb[qi * 3 + 0], qy = cb[qi * 3 + 1], qz = cb[qi * 3 + 2];
    float bd[KNN]; int bi[KNN];
#pragma unroll
    for (int t = 0; t < KNN; ++t) { bd[t] = 3.4e38f; bi[t] = 0; }
    for (int tile = 0; tile < NPTS / 256; ++tile) {
        int base = tile * 256;
        for (int j = threadIdx.x; j < 256 * 3; j += 256) sc[j] = cb[base * 3 + j];
        __syncthreads();
        for (int jj = 0; jj < 256; ++jj) {
            float dx = qx - sc[jj * 3 + 0];
            float dy = qy - sc[jj * 3 + 1];
            float dz = qz - sc[jj * 3 + 2];
            float d = dx * dx + dy * dy + dz * dz;
            int ci = base + jj;
            if (d < bd[KNN - 1]) {
#pragma unroll
                for (int t = KNN - 1; t > 0; --t) {
                    float pd = bd[t - 1]; int pi = bi[t - 1];
                    bool shift = d < pd;
                    bool place = !shift && (d < bd[t]);
                    bd[t] = shift ? pd : (place ? d : bd[t]);
                    bi[t] = shift ? pi : (place ? ci : bi[t]);
                }
                if (d < bd[0]) { bd[0] = d; bi[0] = ci; }
            }
        }
        __syncthreads();
    }
    int* dst = knn + ((size_t)b * NPTS + qi) * KNN;
#pragma unroll
    for (int t = 0; t < KNN; ++t) dst[t] = bi[t];
}

// =================== x = feat@w1+b1 ; q/k/v = x@{wq,wk,wv} (WMMA) ==================
__global__ __launch_bounds__(256) void feat_kernel(const float* __restrict__ feat,
                                                   const float* __restrict__ b1,
                                                   const _Float16* __restrict__ wf,
                                                   float* __restrict__ qo,
                                                   float* __restrict__ xko,
                                                   float* __restrict__ xvo) {
    __shared__ float    sA[8][16][64];
    __shared__ _Float16 sX[8][2][32][16];
    int w = threadIdx.x >> 5, lane = threadIdx.x & 31;
    int row0 = (blockIdx.x * 8 + w) * 16;
    int col0 = lane & 15, hi = lane >> 4;

#pragma unroll
    for (int i = 0; i < 32; ++i) {
        int lin = i * 32 + lane;
        sA[w][lin >> 6][lin & 63] = feat[(size_t)row0 * 64 + lin];
    }
    wait_ds();

    v16h xa[2];
#pragma unroll
    for (int kh = 0; kh < 2; ++kh)
#pragma unroll
        for (int e = 0; e < 16; ++e) {
            int k = 32 * kh + ((e >> 3) << 4) + (hi << 3) + (e & 7);
            xa[kh][e] = (_Float16)sA[w][col0][k];
        }

    v8f cx[4];
#pragma unroll
    for (int nt = 0; nt < 4; ++nt) {
        float bb = b1[col0 + 16 * nt];
#pragma unroll
        for (int v = 0; v < 8; ++v) cx[nt][v] = bb;
#pragma unroll
        for (int kh = 0; kh < 2; ++kh)
            cx[nt] = wmma_f16(xa[kh], loadBfrag(wf, 0, kh, nt, lane), cx[nt]);
    }
    // restage x as A-fragments in LDS
#pragma unroll
    for (int nt = 0; nt < 4; ++nt)
#pragma unroll
        for (int v = 0; v < 8; ++v) {
            int r = v + 8 * hi, c = col0 + 16 * nt;
            sX[w][c >> 5][fragLane(r, c)][fragElem(c)] = (_Float16)cx[nt][v];
        }
    wait_ds();
    v16h qa0 = *(const v16h*)&sX[w][0][lane][0];
    v16h qa1 = *(const v16h*)&sX[w][1][lane][0];

#pragma unroll
    for (int m = 0; m < 3; ++m) {
        float* ob = (m == 0) ? qo : (m == 1) ? xko : xvo;
#pragma unroll
        for (int nt = 0; nt < 4; ++nt) {
            v8f c;
#pragma unroll
            for (int v = 0; v < 8; ++v) c[v] = 0.0f;
            c = wmma_f16(qa0, loadBfrag(wf, 1 + m, 0, nt, lane), c);
            c = wmma_f16(qa1, loadBfrag(wf, 1 + m, 1, nt, lane), c);
#pragma unroll
            for (int v = 0; v < 8; ++v)
                ob[(size_t)(row0 + v + 8 * hi) * 64 + col0 + 16 * nt] = c[v];
        }
    }
}

// ============ fused per-point: pos-enc MLP, attn MLP, softmax, reduce =============
__global__ __launch_bounds__(256) void pair_kernel(
    const float* __restrict__ coords, const int* __restrict__ knn,
    const float* __restrict__ q, const float* __restrict__ xk,
    const float* __restrict__ xv,
    const float* __restrict__ wp1, const float* __restrict__ bp1,
    const float* __restrict__ bp2, const float* __restrict__ bm1,
    const float* __restrict__ bm2,
    const _Float16* __restrict__ wf,
    float* __restrict__ attn_out, float* __restrict__ obuf) {
    __shared__ float    sPos[8][16][64];
    __shared__ _Float16 sFrag[8][2][32][16];
    __shared__ float    sRel[8][16][3];
    __shared__ int      sIdx[8][16];
    __shared__ float    sQ[8][64];

    int w = threadIdx.x >> 5, lane = threadIdx.x & 31;
    int p = blockIdx.x * 8 + w;                 // global point id
    int b = p >> 12, nloc = p & (NPTS - 1);
    const float* cb = coords + (size_t)b * NPTS * 3;
    int r = lane & 15, hi = lane >> 4, cbase = hi * 32, col0 = lane & 15;

    if (lane < 16) {
        int gi = knn[(size_t)p * KNN + lane];
        sIdx[w][lane] = gi;
        sRel[w][lane][0] = cb[nloc * 3 + 0] - cb[gi * 3 + 0];
        sRel[w][lane][1] = cb[nloc * 3 + 1] - cb[gi * 3 + 1];
        sRel[w][lane][2] = cb[nloc * 3 + 2] - cb[gi * 3 + 2];
    }
    sQ[w][lane]      = q[(size_t)p * 64 + lane];
    sQ[w][lane + 32] = q[(size_t)p * 64 + lane + 32];
    wait_ds();

    // h1 = relu(rel @ wp1 + bp1)  (K=3 -> VALU), store directly as f16 A-fragments
    float r0 = sRel[w][r][0], r1 = sRel[w][r][1], r2 = sRel[w][r][2];
#pragma unroll
    for (int j = 0; j < 32; ++j) {
        int c = cbase + j;
        float v = bp1[c] + r0 * wp1[c] + r1 * wp1[64 + c] + r2 * wp1[128 + c];
        v = v > 0.0f ? v : 0.0f;
        sFrag[w][hi][fragLane(r, c)][fragElem(c)] = (_Float16)v;
    }
    wait_ds();
    v16h a0 = *(const v16h*)&sFrag[w][0][lane][0];
    v16h a1 = *(const v16h*)&sFrag[w][1][lane][0];

    // pos = h1 @ wp2 + bp2   (WMMA)
    v8f pos[4];
#pragma unroll
    for (int nt = 0; nt < 4; ++nt) {
        float bb = bp2[col0 + 16 * nt];
#pragma unroll
        for (int v = 0; v < 8; ++v) pos[nt][v] = bb;
        pos[nt] = wmma_f16(a0, loadBfrag(wf, 4, 0, nt, lane), pos[nt]);
        pos[nt] = wmma_f16(a1, loadBfrag(wf, 4, 1, nt, lane), pos[nt]);
    }
#pragma unroll
    for (int nt = 0; nt < 4; ++nt)
#pragma unroll
        for (int v = 0; v < 8; ++v)
            sPos[w][v + 8 * hi][col0 + 16 * nt] = pos[nt][v];
    wait_ds();

    // a_in = q - k_g + pos  -> f16 A-fragments
    int nbr = b * NPTS + sIdx[w][r];
    const float* xkrow = xk + (size_t)nbr * 64;
#pragma unroll
    for (int j = 0; j < 32; ++j) {
        int c = cbase + j;
        float v = sQ[w][c] - xkrow[c] + sPos[w][r][c];
        sFrag[w][hi][fragLane(r, c)][fragElem(c)] = (_Float16)v;
    }
    wait_ds();
    v16h aa0 = *(const v16h*)&sFrag[w][0][lane][0];
    v16h aa1 = *(const v16h*)&sFrag[w][1][lane][0];

    // h2 = relu(a_in @ wm1 + bm1)
    v8f h2[4];
#pragma unroll
    for (int nt = 0; nt < 4; ++nt) {
        float bb = bm1[col0 + 16 * nt];
#pragma unroll
        for (int v = 0; v < 8; ++v) h2[nt][v] = bb;
        h2[nt] = wmma_f16(aa0, loadBfrag(wf, 5, 0, nt, lane), h2[nt]);
        h2[nt] = wmma_f16(aa1, loadBfrag(wf, 5, 1, nt, lane), h2[nt]);
    }
#pragma unroll
    for (int nt = 0; nt < 4; ++nt)
#pragma unroll
        for (int v = 0; v < 8; ++v) {
            float x = h2[nt][v]; x = x > 0.0f ? x : 0.0f;
            int rr = v + 8 * hi, cc = col0 + 16 * nt;
            sFrag[w][cc >> 5][fragLane(rr, cc)][fragElem(cc)] = (_Float16)x;
        }
    wait_ds();
    v16h ha0 = *(const v16h*)&sFrag[w][0][lane][0];
    v16h ha1 = *(const v16h*)&sFrag[w][1][lane][0];

    // att = h2 @ wm2 + bm2
    v8f att[4];
#pragma unroll
    for (int nt = 0; nt < 4; ++nt) {
        float bb = bm2[col0 + 16 * nt];
#pragma unroll
        for (int v = 0; v < 8; ++v) att[nt][v] = bb;
        att[nt] = wmma_f16(ha0, loadBfrag(wf, 6, 0, nt, lane), att[nt]);
        att[nt] = wmma_f16(ha1, loadBfrag(wf, 6, 1, nt, lane), att[nt]);
    }

    // softmax over the 16 neighbor rows (per column): 8 rows per lane + lane^16
    const float inv_s = 0.125f;   // 1/sqrt(64)
#pragma unroll
    for (int nt = 0; nt < 4; ++nt) {
        float m = -3.4e38f;
#pragma unroll
        for (int v = 0; v < 8; ++v) { att[nt][v] *= inv_s; m = fmaxf(m, att[nt][v]); }
        m = fmaxf(m, __shfl_xor(m, 16, 32));
        float s = 0.0f;
#pragma unroll
        for (int v = 0; v < 8; ++v) { float e = __expf(att[nt][v] - m); att[nt][v] = e; s += e; }
        s += __shfl_xor(s, 16, 32);
        float rs = 1.0f / s;
#pragma unroll
        for (int v = 0; v < 8; ++v) att[nt][v] *= rs;
    }

    // write attn + reduce sum_k attn * (v_g + pos)
    int nbv[8];
#pragma unroll
    for (int v = 0; v < 8; ++v) nbv[v] = b * NPTS + sIdx[w][v + 8 * hi];
    float* aout = attn_out + (size_t)p * KNN * 64;
#pragma unroll
    for (int nt = 0; nt < 4; ++nt) {
        int c = col0 + 16 * nt;
        float acc = 0.0f;
#pragma unroll
        for (int v = 0; v < 8; ++v) {
            int rr = v + 8 * hi;
            float a = att[nt][v];
            aout[rr * 64 + c] = a;
            float ve = xv[(size_t)nbv[v] * 64 + c] + sPos[w][rr][c];
            acc += a * ve;
        }
        acc += __shfl_xor(acc, 16, 32);
        if (hi == 0) obuf[(size_t)p * 64 + c] = acc;
    }
}

// =================== out = o @ w2 + b2 + features (WMMA) ==========================
__global__ __launch_bounds__(256) void final_kernel(const float* __restrict__ ob,
                                                    const float* __restrict__ feat,
                                                    const float* __restrict__ b2,
                                                    const _Float16* __restrict__ wf,
                                                    float* __restrict__ out) {
    __shared__ float sA[8][16][64];
    int w = threadIdx.x >> 5, lane = threadIdx.x & 31;
    int row0 = (blockIdx.x * 8 + w) * 16;
    int col0 = lane & 15, hi = lane >> 4;
#pragma unroll
    for (int i = 0; i < 32; ++i) {
        int lin = i * 32 + lane;
        sA[w][lin >> 6][lin & 63] = ob[(size_t)row0 * 64 + lin];
    }
    wait_ds();
    v16h oa[2];
#pragma unroll
    for (int kh = 0; kh < 2; ++kh)
#pragma unroll
        for (int e = 0; e < 16; ++e) {
            int k = 32 * kh + ((e >> 3) << 4) + (hi << 3) + (e & 7);
            oa[kh][e] = (_Float16)sA[w][col0][k];
        }
#pragma unroll
    for (int nt = 0; nt < 4; ++nt) {
        float bb = b2[col0 + 16 * nt];
        v8f c;
#pragma unroll
        for (int v = 0; v < 8; ++v) c[v] = bb;
        c = wmma_f16(oa[0], loadBfrag(wf, 7, 0, nt, lane), c);
        c = wmma_f16(oa[1], loadBfrag(wf, 7, 1, nt, lane), c);
#pragma unroll
        for (int v = 0; v < 8; ++v) {
            size_t idx = (size_t)(row0 + v + 8 * hi) * 64 + col0 + 16 * nt;
            out[idx] = c[v] + feat[idx];
        }
    }
}

extern "C" void kernel_launch(void* const* d_in, const int* in_sizes, int n_in,
                              void* d_out, int out_size, void* d_ws, size_t ws_size,
                              hipStream_t stream) {
    const float* coords = (const float*)d_in[0];
    const float* feat   = (const float*)d_in[1];
    const float* w1  = (const float*)d_in[2];
    const float* b1  = (const float*)d_in[3];
    const float* w2  = (const float*)d_in[4];
    const float* b2  = (const float*)d_in[5];
    const float* wq  = (const float*)d_in[6];
    const float* wk  = (const float*)d_in[7];
    const float* wv  = (const float*)d_in[8];
    const float* wm1 = (const float*)d_in[9];
    const float* bm1 = (const float*)d_in[10];
    const float* wm2 = (const float*)d_in[11];
    const float* bm2 = (const float*)d_in[12];
    const float* wp1 = (const float*)d_in[13];
    const float* bp1 = (const float*)d_in[14];
    const float* wp2 = (const float*)d_in[15];
    const float* bp2 = (const float*)d_in[16];

    float* out      = (float*)d_out;
    float* attn_out = out + (size_t)BATCH * NPTS * 64;

    char* ws = (char*)d_ws;
    _Float16* wf = (_Float16*)ws;                           // 64 KB (8 matrices)
    int*   knn = (int*)(ws + 65536);                        // 1 MB
    float* qb  = (float*)(ws + 65536 + (size_t)BATCH * NPTS * KNN * 4);
    float* xkb = qb  + (size_t)BATCH * NPTS * 64;
    float* xvb = xkb + (size_t)BATCH * NPTS * 64;
    float* ob  = xvb + (size_t)BATCH * NPTS * 64;

    prep_kernel<<<8, 256, 0, stream>>>(w1, wq, wk, wv, wp2, wm1, wm2, w2, wf);
    knn_kernel<<<BATCH * (NPTS / 256), 256, 0, stream>>>(coords, knn);
    feat_kernel<<<BATCH * NPTS / 128, 256, 0, stream>>>(feat, b1, wf, qb, xkb, xvb);
    pair_kernel<<<BATCH * NPTS / 8, 256, 0, stream>>>(coords, knn, qb, xkb, xvb,
                                                      wp1, bp1, bp2, bm1, bm2,
                                                      wf, attn_out, ob);
    final_kernel<<<BATCH * NPTS / 128, 256, 0, stream>>>(ob, feat, b2, wf, out);
}
